// sbvr_69569880260957
// MI455X (gfx1250) — matrix operand order repaired and checked by reading, not compiled, not run on hardware
//
#include <hip/hip_runtime.h>
#include <stdint.h>

// decoded[g, l] = sum_s coeff_cache[coeff_idx[g], s] * ((bvr[g,s] >> l) & 1)
// G = 4M groups, S = 4, L = 16. Memory-bound (~336 MB/call -> ~14us @ 23.3TB/s).
//
// WMMA mapping: tile = 16 groups -> one 16x16 f32 output tile.
//   K packs (group-in-block, sum) pairs; WMMA#1 covers groups 0..7,
//   WMMA#2 covers groups 8..15 (A block-diagonal coeff, B = exact 0/1 bits).
//   coeff split into f16 hi+lo -> 4x v_wmma_f32_16x16x32_f16, f32 accumulate.
// 2 tiles per loop iteration: one async LDS copy stages 32 groups with all
// 32 lanes; uniform bases readfirstlane'd so addressing is SALU + SADDR-form.

typedef __attribute__((ext_vector_type(16))) _Float16 v16h;
typedef __attribute__((ext_vector_type(8)))  float    v8f;

#define SBVR_NUM_GROUPS (8192 * 8192 / 16)       /* 4,194,304 */
#define SBVR_NUM_TILES  (SBVR_NUM_GROUPS / 16)   /* 262,144 */
#define SBVR_NUM_BLOCKS (SBVR_NUM_TILES / 2)     /* 131,072 (2 tiles/block) */

union VH { v16h h; uint32_t u[8]; };             // WMMA operand as 8 dwords
union HP { _Float16 f[2]; uint32_t u; };         // packed f16 pair

__device__ __forceinline__ unsigned ucomp4(int4 v, int i) {
  return (unsigned)(i==0 ? v.x : i==1 ? v.y : i==2 ? v.z : v.w);
}
__device__ __forceinline__ float fcomp4(float4 v, int i) {
  return i==0 ? v.x : i==1 ? v.y : i==2 ? v.z : v.w;
}

// One 16-group tile: coeff split/pack, A/B build, 4 chained WMMAs, 8 stores.
__device__ __forceinline__ void sbvr_tile(const float4 c,
                                          const int4* __restrict__ slice16,
                                          float* __restrict__ outb,  // uniform
                                          const int m, const int hi)
{
  // coeff -> f16 hi+lo, packed into dword pairs (s0,s1)/(s2,s3).
  uint32_t pkh[2], pkl[2];
#pragma unroll
  for (int p = 0; p < 2; ++p) {
    HP h, l;
#pragma unroll
    for (int q = 0; q < 2; ++q) {
      const float cf = fcomp4(c, 2 * p + q);
      const _Float16 hh = (_Float16)cf;
      h.f[q] = hh;
      l.f[q] = (_Float16)(cf - (float)hh);
    }
    pkh[p] = h.u;
    pkl[p] = l.u;
  }

  // A operands (block-diagonal coeff), 8 dword selects each.
  // A k order chosen here (B matches): slot j holds K = j + 16*hi; dword r
  // covers slots (2r,2r+1) -> one K>>2 group, sum pair (0,1) or (2,3).
  // WMMA#1: nonzero iff K>>2 == m ; WMMA#2: nonzero iff (K>>2)+8 == m.
  VH a1h, a1l, a2h, a2l;
#pragma unroll
  for (int r = 0; r < 8; ++r) {
    const int j0 = 2 * r;
    const int k  = j0 + 16 * hi;
    const int g  = k >> 2;
    const int sp = (j0 & 3) >> 1;
    const bool c1 = (g == m);
    const bool c2 = (g + 8 == m);
    a1h.u[r] = c1 ? pkh[sp] : 0u;
    a1l.u[r] = c1 ? pkl[sp] : 0u;
    a2h.u[r] = c2 ? pkh[sp] : 0u;
    a2l.u[r] = c2 ? pkl[sp] : 0u;
  }

  // Gather the 8 groups this lane needs from the staged tile.
  int4 W1[4], W2[4];
#pragma unroll
  for (int t = 0; t < 4; ++t) {
    W1[t] = slice16[4 * hi + t];        // groups 4*hi .. 4*hi+3    (k-block 1)
    W2[t] = slice16[8 + 4 * hi + t];    // groups 8+4*hi .. 11+4*hi (k-block 2)
  }

  // B operands: slot j holds K = j + 16*hi, column n = m.
  // Two bits -> one dword: (bfe, lshl, and_or, mul 0x3C00) = f16 {0,1} pair.
  VH b1, b2;
#pragma unroll
  for (int r = 0; r < 8; ++r) {
    const int t  = r >> 1;
    const int s0 = (2 * r) & 3;
    const uint32_t q1 = ((ucomp4(W1[t], s0) >> m) & 1u) |
                        (((ucomp4(W1[t], s0 + 1) >> m) & 1u) << 16);
    const uint32_t q2 = ((ucomp4(W2[t], s0) >> m) & 1u) |
                        (((ucomp4(W2[t], s0 + 1) >> m) & 1u) << 16);
    b1.u[r] = q1 * 0x3C00u;             // f16 1.0 per set bit
    b2.u[r] = q2 * 0x3C00u;
  }

  // 4 WMMAs: (hi + lo coeff) x (two k-blocks), f32 accumulate.
  v8f acc = {};
  acc = __builtin_amdgcn_wmma_f32_16x16x32_f16(false, a1h.h, false, b1.h, (short)0, acc, false, false);
  acc = __builtin_amdgcn_wmma_f32_16x16x32_f16(false, a2h.h, false, b2.h, (short)0, acc, false, false);
  acc = __builtin_amdgcn_wmma_f32_16x16x32_f16(false, a1l.h, false, b1.h, (short)0, acc, false, false);
  acc = __builtin_amdgcn_wmma_f32_16x16x32_f16(false, a2l.h, false, b2.h, (short)0, acc, false, false);

  // Store D directly (no VALU touch after WMMA -> no hazard nops):
  // VGPR r -> row r (lanes 0-15) / r+8 (lanes 16-31), column m.
#pragma unroll
  for (int r = 0; r < 8; ++r)
    outb[(size_t)(r + 8 * hi) * 16 + m] = acc[r];
}

__global__ __launch_bounds__(256) void sbvr_decode_wmma(
    const float4* __restrict__ cache4,     // [65536] float4 (coeff_cache rows)
    const int*    __restrict__ coeff_idx,  // [NUM_GROUPS]
    const int4*   __restrict__ bvr4,       // [NUM_GROUPS] 4 packed 16-bit words
    float*        __restrict__ out)        // [NUM_GROUPS * 16]
{
  __shared__ int4 lds_bvr[8 * 32];         // 32 groups (2 tiles) per wave

  const int lane = threadIdx.x & 31;
  const int wave = threadIdx.x >> 5;
  const int m    = lane & 15;              // A row / B,D column index
  const int hi   = lane >> 4;              // half-wave selector
  const uint32_t waves_per_blk = blockDim.x >> 5;
  const uint32_t stride        = gridDim.x * waves_per_blk;   // total waves
  int4* slice = &lds_bvr[wave * 32];

  // Each lane stages one group of the 2-tile block into its own slot.
  const uint32_t lds_addr = (uint32_t)(uintptr_t)(const void*)&slice[lane];
  const uint32_t bvr_voff = (uint32_t)lane * 16u;   // byte offset in block

  // Wave-uniform block index, made provably uniform -> SALU induction,
  // SADDR-form memory addressing everywhere.
  uint32_t blk = (uint32_t)__builtin_amdgcn_readfirstlane(
      (int)(blockIdx.x * waves_per_blk + wave));

  for (; blk < SBVR_NUM_BLOCKS; blk += stride) {
    // ---- CDNA5 async copy: 32 x 16B of bvr straight into LDS (ASYNCcnt).
    const char* bvr_base = (const char*)(bvr4 + (size_t)blk * 32);
    asm volatile("global_load_async_to_lds_b128 %0, %1, %2"
                 :: "v"(lds_addr), "v"(bvr_voff), "s"(bvr_base)
                 : "memory");

    // Speculative prefetch of the next block (global_prefetch_b8):
    // TH=0 speculative => silently dropped if past the mapped range.
    __builtin_prefetch((const void*)(bvr_base + (size_t)stride * 512 + bvr_voff), 0, 1);

    // ---- Both tiles' coeff gathers issued before the async wait.
    const int* idxb = coeff_idx + (size_t)blk * 32;
    const int ci0 = idxb[m];
    const int ci1 = idxb[16 + m];
    const float4 c0 = cache4[ci0];
    const float4 c1 = cache4[ci1];

    asm volatile("s_wait_asynccnt 0" ::: "memory");

    float* outb = out + (size_t)blk * 512;
    sbvr_tile(c0, slice,      outb,       m, hi);   // tile 0: groups 0..15
    sbvr_tile(c1, slice + 16, outb + 256, m, hi);   // tile 1: groups 16..31
  }
}

extern "C" void kernel_launch(void* const* d_in, const int* in_sizes, int n_in,
                              void* d_out, int out_size, void* d_ws, size_t ws_size,
                              hipStream_t stream) {
  const float4* cache4 = (const float4*)d_in[0];  // coeff_cache [65536,4] f32
  const int*    cidx   = (const int*)d_in[1];     // coeff_idx   [4194304] i32
  const int4*   bvr4   = (const int4*)d_in[2];    // bvr         [4194304,4] i32
  float*        out    = (float*)d_out;           // [8192*8192] f32

  dim3 block(256);   // 8 wave32 waves per block
  dim3 grid(4096);   // 32768 waves -> exactly 4 blocks (8 tiles) per wave
  hipLaunchKernelGGL(sbvr_decode_wmma, grid, block, 0, stream,
                     cache4, cidx, bvr4, out);
}
